// SelectModel_30331059044670
// MI455X (gfx1250) — compile-verified, matrix operand order checked by reading, not run on hardware
//
#include <hip/hip_runtime.h>
#include <hip/hip_bf16.h>
#include <math.h>

typedef float v2f  __attribute__((ext_vector_type(2)));
typedef float v8f  __attribute__((ext_vector_type(8)));
typedef __bf16 bf16x16 __attribute__((ext_vector_type(16)));

#define N_REGIONS 2048
#define D_IN      3
#define D         100     // embed / lstm / out dim
#define DP        112     // D padded to 16 (for cov / racc)
#define DK        128     // D padded to 32 (bf16 WMMA K)
#define G4        400     // 4 gates * 100
#define H1        1000
#define H1P       1008    // H1 padded to 16 (N tiles)
#define H1K       1024    // H1 padded to 32 (bf16 WMMA K)

// ---- workspace layout (dword offsets) ----
#define OFF_G0     0                            // 400 floats (rounded to 512)
#define OFF_WIHP   512                          // 4 kb * 400 * 16  = 25600 dwords
#define OFF_W1P    (OFF_WIHP + 4*G4*16)         // 4 kb * 1008 * 16 = 64512
#define OFF_W2P    (OFF_W1P + 4*H1P*16)         // 32 kb * 112 * 16 = 57344
#define OFF_RACC   (OFF_W2P + 32*DP*16)         // 2048 * 112 floats = 229376
// total ~377,344 dwords (~1.5 MB)

// ---- LDS layout for main kernel (float offsets, odd-mod-64 strides) ----
#define LDE   129   // e/h row stride (128 cols + 1)
#define LDG   401   // gates row stride
#define LDT   1025  // t row stride (1024 cols + 1)
#define SM_E    0
#define SM_G    (SM_E + 16*LDE)     // 2064
#define SM_H    (SM_G + 16*LDG)     // 8480
#define SM_T    (SM_H + 16*LDE)     // 10544
#define SM_RID  (SM_T + 16*LDT)     // 26944
#define SM_FLOATS (SM_RID + 16)     // 26960 floats = 107840 bytes

__device__ __forceinline__ float sigf(float x) { return 1.0f / (1.0f + __expf(-x)); }

__device__ __forceinline__ unsigned short f2bf(float f) {
    union { float f; unsigned u; } x; x.f = f;
    unsigned r = x.u + 0x7fffu + ((x.u >> 16) & 1u);   // RTNE
    return (unsigned short)(r >> 16);
}

// Build an A fragment (16x32 bf16, one k-block) from an f32 LDS row.
// kbase(v,half) = 16*(v>>2) + 8*half + 2*(v&3)
__device__ __forceinline__ bf16x16 a_frag(const float* __restrict__ row, int kb32, int half) {
    bf16x16 a;
    const int k0 = kb32 + 8 * half;        // v = 0..3 run
    const int k1 = kb32 + 16 + 8 * half;   // v = 4..7 run
    #pragma unroll
    for (int v = 0; v < 4; ++v) {
        a[2*v]     = (__bf16)row[k0 + 2*v];
        a[2*v + 1] = (__bf16)row[k0 + 2*v + 1];
        a[8 + 2*v]     = (__bf16)row[k1 + 2*v];
        a[8 + 2*v + 1] = (__bf16)row[k1 + 2*v + 1];
    }
    return a;
}

union BFrag { bf16x16 v; unsigned u[8]; };

// g0[n] = bih[n] + bhh[n] + sum_k h0[k]*Whh[n][k]
__global__ void prep_g0_kernel(const float* __restrict__ bih, const float* __restrict__ bhh,
                               const float* __restrict__ h0, const float* __restrict__ Whh,
                               float* __restrict__ g0) {
    int n = blockIdx.x * blockDim.x + threadIdx.x;
    if (n >= G4) return;
    float s = bih[n] + bhh[n];
    const float* w = Whh + n * D;
    #pragma unroll 4
    for (int k = 0; k < D; ++k) s += h0[k] * w[k];
    g0[n] = s;
}

// Pack W (row-major [Nin][Kin], used as B[k][n] = W[n][k]) into bf16 WMMA B-fragment
// order: out[((kb*Np + n)*16) + half*8 + v] = pack(bf16 B[k][n], bf16 B[k+1][n]),
// k = kb*32 + 16*(v>>2) + 8*half + 2*(v&3).  Zero-pads k>=Kin and n>=Nin.
__global__ void pack_b_kernel(const float* __restrict__ in, unsigned* __restrict__ out,
                              int Kin, int Nin, int KB, int Np) {
    int total = KB * Np * 16;
    for (int idx = blockIdx.x * blockDim.x + threadIdx.x; idx < total; idx += gridDim.x * blockDim.x) {
        int p  = idx & 15;
        int n  = (idx >> 4) % Np;
        int kb = idx / (16 * Np);
        int v = p & 7, half = p >> 3;
        int k = kb * 32 + 16 * (v >> 2) + 8 * half + 2 * (v & 3);
        float lo = 0.0f, hi = 0.0f;
        if (n < Nin) {
            if (k     < Kin) lo = in[n * Kin + k];
            if (k + 1 < Kin) hi = in[n * Kin + k + 1];
        }
        out[idx] = ((unsigned)f2bf(hi) << 16) | (unsigned)f2bf(lo);
    }
}

__global__ void zero_kernel(float* __restrict__ p, int count) {
    for (int i = blockIdx.x * blockDim.x + threadIdx.x; i < count; i += gridDim.x * blockDim.x)
        p[i] = 0.0f;
}

// Fused per-route pipeline: embed -> gates GEMM -> LSTM -> FC1+ReLU -> FC2 -> region atomic sum
__global__ void __launch_bounds__(128)
routes_fused_kernel(const float* __restrict__ routes, const int* __restrict__ rids,
                    const float* __restrict__ W0, const float* __restrict__ b0,
                    const float* __restrict__ c0,
                    const float* __restrict__ b1, const float* __restrict__ b2,
                    const unsigned* __restrict__ ws_u, const float* __restrict__ ws_f,
                    float* __restrict__ racc, int N) {
    extern __shared__ float sm[];
    int* ridS = (int*)&sm[SM_RID];

    const int tid  = threadIdx.x;
    const int lane = tid & 31;
    const int wv   = tid >> 5;           // 4 waves
    const int col  = lane & 15;
    const int half = lane >> 4;
    const int base = blockIdx.x * 16;

    const float*    g0   = ws_f + OFF_G0;
    const unsigned* WihP = ws_u + OFF_WIHP;   // [4][G4][16]
    const unsigned* W1P  = ws_u + OFF_W1P;    // [4][H1P][16]
    const unsigned* W2P  = ws_u + OFF_W2P;    // [32][DP][16]

    // ---- stage 0: region ids; embed e = routes @ W0^T + b0 -> LDS [16][DK]; zero t pad cols ----
    if (tid < 16) {
        int r = base + tid;
        ridS[tid] = (r < N) ? rids[r] : 0;
    }
    for (int idx = tid; idx < 16 * 16; idx += 128)           // t cols 1008..1023 = 0
        sm[SM_T + (idx >> 4) * LDT + H1P + (idx & 15)] = 0.0f;
    for (int idx = tid; idx < 16 * DK; idx += 128) {
        int r = idx / DK, c = idx % DK;
        int route = base + r;
        float v = 0.0f;
        if (route < N && c < D) {
            const float* x = routes + route * D_IN;
            const float* w = W0 + c * D_IN;
            v = b0[c] + x[0] * w[0] + x[1] * w[1] + x[2] * w[2];
        }
        sm[SM_E + r * LDE + c] = v;
    }
    __syncthreads();

    // ---- stage 1: gates[16][400] = e @ Wih^T (+ g0); bf16 WMMA 16x16x32, K=128 ----
    for (int t = wv; t < G4 / 16; t += 4) {
        int n0 = t * 16;
        v8f acc = {};
        #pragma unroll
        for (int kb = 0; kb < 4; ++kb) {
            bf16x16 a = a_frag(&sm[SM_E + col * LDE], kb * 32, half);
            BFrag b;
            const unsigned* bp = WihP + ((kb * G4 + n0 + col) << 4) + half * 8;
            #pragma unroll
            for (int i = 0; i < 8; ++i) b.u[i] = bp[i];
            acc = __builtin_amdgcn_wmma_f32_16x16x32_bf16(false, a, false, b.v, (short)0, acc, false, false);
        }
        #pragma unroll
        for (int v = 0; v < 8; ++v) {
            int M = v + 8 * half;
            sm[SM_G + M * LDG + n0 + col] = acc[v] + g0[n0 + col];
        }
    }
    __syncthreads();

    // ---- stage 2: LSTM pointwise -> h [16][DK] ----
    for (int idx = tid; idx < 16 * DK; idx += 128) {
        int r = idx / DK, c = idx % DK;
        float hv = 0.0f;
        if (c < D) {
            float ig = sm[SM_G + r * LDG + c];
            float fg = sm[SM_G + r * LDG + D + c];
            float gg = sm[SM_G + r * LDG + 2 * D + c];
            float og = sm[SM_G + r * LDG + 3 * D + c];
            float cc = sigf(fg) * c0[c] + sigf(ig) * tanhf(gg);
            hv = sigf(og) * tanhf(cc);
        }
        sm[SM_H + r * LDE + c] = hv;
    }
    __syncthreads();

    // ---- stage 3: t[16][1008] = relu(h @ W1^T + b1); bf16 WMMA, K=128 ----
    for (int t = wv; t < H1P / 16; t += 4) {
        int n0 = t * 16;
        v8f acc = {};
        #pragma unroll
        for (int kb = 0; kb < 4; ++kb) {
            bf16x16 a = a_frag(&sm[SM_H + col * LDE], kb * 32, half);
            BFrag b;
            const unsigned* bp = W1P + ((kb * H1P + n0 + col) << 4) + half * 8;
            #pragma unroll
            for (int i = 0; i < 8; ++i) b.u[i] = bp[i];
            acc = __builtin_amdgcn_wmma_f32_16x16x32_bf16(false, a, false, b.v, (short)0, acc, false, false);
        }
        int n = n0 + col;
        #pragma unroll
        for (int v = 0; v < 8; ++v) {
            int M = v + 8 * half;
            float val = 0.0f;
            if (n < H1) { val = acc[v] + b1[n]; val = val > 0.0f ? val : 0.0f; }
            sm[SM_T + M * LDT + n] = val;
        }
    }
    __syncthreads();

    // ---- stage 4: hx[16][100] = t @ W2^T + b2; bf16 WMMA, K=1024; atomic segment-sum ----
    for (int t = wv; t < DP / 16; t += 4) {
        int n0 = t * 16;
        v8f acc = {};
        for (int kb = 0; kb < 32; ++kb) {
            bf16x16 a = a_frag(&sm[SM_T + col * LDT], kb * 32, half);
            BFrag b;
            const unsigned* bp = W2P + ((kb * DP + n0 + col) << 4) + half * 8;
            #pragma unroll
            for (int i = 0; i < 8; ++i) b.u[i] = bp[i];
            acc = __builtin_amdgcn_wmma_f32_16x16x32_bf16(false, a, false, b.v, (short)0, acc, false, false);
        }
        int n = n0 + col;
        if (n < D) {
            #pragma unroll
            for (int v = 0; v < 8; ++v) {
                int M = v + 8 * half;
                int route = base + M;
                if (route < N)
                    atomicAdd(&racc[ridS[M] * DP + n], acc[v] + b2[n]);
            }
        }
    }
}

// L2-normalize each region row in place (pad cols stay zero)
__global__ void __launch_bounds__(128)
normalize_kernel(float* __restrict__ racc) {
    __shared__ float red[128];
    int r = blockIdx.x;
    int tid = threadIdx.x;
    float v = 0.0f;
    if (tid < DP) v = racc[r * DP + tid];
    red[tid] = v * v;
    __syncthreads();
    for (int s = 64; s > 0; s >>= 1) {
        if (tid < s) red[tid] += red[tid + s];
        __syncthreads();
    }
    float nrm = sqrtf(red[0]);
    if (tid < DP) racc[r * DP + tid] = v / nrm;
}

// cov = emb @ emb^T [2048 x 2048], diagonal = -inf. Exact f32 WMMA 16x16x4, K=112.
__global__ void __launch_bounds__(256)
cov_kernel(const float* __restrict__ emb, float* __restrict__ cov) {
    const int lane = threadIdx.x & 31;
    const int wv   = threadIdx.x >> 5;    // 8 waves
    const int col  = lane & 15;
    const int half = lane >> 4;

    int tile = blockIdx.x * 8 + wv;       // 128x128 tiles total
    int i0 = (tile >> 7) * 16;
    int j0 = (tile & 127) * 16;

    v8f acc = {};
    #pragma unroll 4
    for (int k = 0; k < DP; k += 4) {
        v2f a, b;
        int ka = k + 2 * half;
        a.x = emb[(i0 + col) * DP + ka];
        a.y = emb[(i0 + col) * DP + ka + 1];
        b.x = emb[(j0 + col) * DP + ka];
        b.y = emb[(j0 + col) * DP + ka + 1];
        acc = __builtin_amdgcn_wmma_f32_16x16x4_f32(false, a, false, b, (short)0, acc, false, false);
    }
    #pragma unroll
    for (int v = 0; v < 8; ++v) {
        int M = v + 8 * half;
        int i = i0 + M, j = j0 + col;
        cov[i * N_REGIONS + j] = (i == j) ? -__builtin_inff() : acc[v];
    }
}

extern "C" void kernel_launch(void* const* d_in, const int* in_sizes, int n_in,
                              void* d_out, int out_size, void* d_ws, size_t ws_size,
                              hipStream_t stream) {
    const float* routes = (const float*)d_in[0];
    const int*   rids   = (const int*)d_in[1];
    const float* W0     = (const float*)d_in[2];
    const float* b0     = (const float*)d_in[3];
    const float* Wih    = (const float*)d_in[4];
    const float* bih    = (const float*)d_in[5];
    const float* Whh    = (const float*)d_in[6];
    const float* bhh    = (const float*)d_in[7];
    const float* h0     = (const float*)d_in[8];
    const float* c0     = (const float*)d_in[9];
    const float* W1     = (const float*)d_in[10];
    const float* b1     = (const float*)d_in[11];
    const float* W2     = (const float*)d_in[12];
    const float* b2     = (const float*)d_in[13];
    float*    out  = (float*)d_out;
    float*    wsf  = (float*)d_ws;
    unsigned* wsu  = (unsigned*)d_ws;

    const int N = in_sizes[0] / D_IN;   // 200000

    // prep: gate bias fold, bf16 B-fragment weight packs, zero accumulator
    prep_g0_kernel<<<(G4 + 127) / 128, 128, 0, stream>>>(bih, bhh, h0, Whh, wsf + OFF_G0);
    pack_b_kernel<<<256, 256, 0, stream>>>(Wih, wsu + OFF_WIHP, D,  G4, 4,  G4);
    pack_b_kernel<<<256, 256, 0, stream>>>(W1,  wsu + OFF_W1P,  D,  H1, 4,  H1P);
    pack_b_kernel<<<256, 256, 0, stream>>>(W2,  wsu + OFF_W2P,  H1, D,  32, DP);
    zero_kernel<<<256, 256, 0, stream>>>(wsf + OFF_RACC, N_REGIONS * DP);

    // fused per-route pipeline (16 routes / block, 4 waves)
    int nblk = (N + 15) / 16;
    routes_fused_kernel<<<nblk, 128, SM_FLOATS * sizeof(float), stream>>>(
        routes, rids, W0, b0, c0, b1, b2, wsu, wsf, wsf + OFF_RACC, N);

    // normalize rows, then cosine-similarity GEMM with -inf diagonal
    normalize_kernel<<<N_REGIONS, 128, 0, stream>>>(wsf + OFF_RACC);
    cov_kernel<<<(N_REGIONS / 16) * (N_REGIONS / 16) / 8, 256, 0, stream>>>(wsf + OFF_RACC, out);
}
